// _SelfAttentionMask_13391708029484
// MI455X (gfx1250) — compile-verified, matrix operand order checked by reading, not run on hardware
//
#include <hip/hip_runtime.h>
#include <hip/hip_bf16.h>

// CDNA5 wave32 WMMA types
typedef __attribute__((ext_vector_type(16))) _Float16 v16h;
typedef __attribute__((ext_vector_type(8)))  float    v8f;

#define B_   16
#define C_   128
#define N_   1024
#define KO_  512    // CHOICES * OUT_C
#define PR_  768    // stacked projection rows (Wq 128 | Wk 128 | Wv 512)

// WMMA f16 operand: VGPR v (lane half h) covers K pair ((v<4)?0:16)+h*8+(v&3)*2.
// => per lane, v=0..3 is 16 contiguous bytes at byte offset h*16, v=4..7 at h*16+32.
// So one operand = two b128 loads per lane.
union ABu { v16h v; uint4 q[2]; unsigned u[8]; _Float16 h[16]; };
union Cu  { v8f  v; float f[8]; };
union PKu { uint4 q; _Float16 h[8]; };

// ---------- 1) x [B,C,N] fp32 -> xT [B,N,C] f16, LDS-tiled transpose ----------
__global__ void k_convert_x(const float* __restrict__ x, _Float16* __restrict__ xT) {
    __shared__ float tile[32][33];
    int blk   = blockIdx.x;
    int ntile = blk & 31;            // N/32
    int ctile = (blk >> 5) & 3;      // C/32
    int b     = blk >> 7;
    int tx = threadIdx.x;            // 0..31
    int ty = threadIdx.y;            // 0..7
#pragma unroll
    for (int s = 0; s < 4; ++s) {
        int c = ctile * 32 + ty + s * 8;
        int n = ntile * 32 + tx;
        tile[ty + s * 8][tx] = x[((size_t)(b * C_ + c)) * N_ + n];   // coalesced read
    }
    __syncthreads();
#pragma unroll
    for (int s = 0; s < 4; ++s) {
        int n = ntile * 32 + ty + s * 8;
        int c = ctile * 32 + tx;
        xT[((size_t)(b * N_ + n)) * C_ + c] = (_Float16)tile[tx][ty + s * 8]; // coalesced write
    }
}

// ---------- 2) stack Wq|Wk|Wv -> W16 [768,128] f16 ----------
__global__ void k_convert_w(const float* __restrict__ Wq, const float* __restrict__ Wk,
                            const float* __restrict__ Wv, _Float16* __restrict__ W16) {
    int idx = blockIdx.x * 256 + threadIdx.x;      // [0, 768*128)
    int o = idx >> 7;
    float v = (o < 128) ? Wq[idx] : (o < 256) ? Wk[idx - 128 * 128] : Wv[idx - 256 * 128];
    W16[idx] = (_Float16)v;
}

// ---------- 3) projection GEMM: P[o,n] = sum_c W16[o,c] * x2[c,n], one wave per 16x16 tile ----------
__global__ void k_proj(const _Float16* __restrict__ W16, const _Float16* __restrict__ xT,
                       _Float16* __restrict__ fT, _Float16* __restrict__ gT,
                       _Float16* __restrict__ hv) {
    __shared__ __align__(16) _Float16 lds[8][256];   // per-wave 16x16 transpose tile
    int lane  = threadIdx.x & 31;
    int wslot = (threadIdx.x >> 5) & 7;
    int wave  = (blockIdx.x * blockDim.x + threadIdx.x) >> 5;
    int ntile = wave & 63;
    int tmp   = wave >> 6;
    int otile = tmp % 48;
    int b     = tmp / 48;
    int m = lane & 15, half = lane >> 4;

    const uint4* W4 = (const uint4*)W16;   // row stride: 256 B = 16 uint4
    const uint4* X4 = (const uint4*)xT;
    int orow = otile * 16 + m;
    int ncol = ntile * 16 + m;
    size_t abase = (size_t)orow * 16 + half;
    size_t bbase = ((size_t)(b * N_ + ncol)) * 16 + half;

    Cu c;
#pragma unroll
    for (int r = 0; r < 8; ++r) c.f[r] = 0.f;

#pragma unroll
    for (int chunk = 0; chunk < 4; ++chunk) {
        ABu a, bb;
        a.q[0]  = W4[abase + chunk * 4];
        a.q[1]  = W4[abase + chunk * 4 + 2];
        bb.q[0] = X4[bbase + chunk * 4];
        bb.q[1] = X4[bbase + chunk * 4 + 2];
        c.v = __builtin_amdgcn_wmma_f32_16x16x32_f16(false, a.v, false, bb.v,
                                                     (short)0, c.v, false, false);
    }

    // Routing is wave-uniform (region bounds are tile-aligned): make it scalar.
    int osel = __builtin_amdgcn_readfirstlane(otile);
    if (osel < 16) {
        // f / g, transposed [b,n,c]: per lane the 8 outputs are 8 consecutive f16 -> one b128 store
        PKu pk;
#pragma unroll
        for (int r = 0; r < 8; ++r) pk.h[r] = (_Float16)c.f[r];
        _Float16* dst = (osel < 8) ? fT : gT;
        int col0 = (otile - ((osel < 8) ? 0 : 8)) * 16 + half * 8;
        *(uint4*)&dst[(size_t)(b * N_ + ncol) * C_ + col0] = pk.q;
    } else {
        // hv, row-major [b,512,n]: transpose via per-wave LDS tile, then one b128 store per lane
#pragma unroll
        for (int r = 0; r < 8; ++r)
            lds[wslot][(r + 8 * half) * 16 + m] = (_Float16)c.f[r];
        __builtin_amdgcn_wave_barrier();      // DS ops are in-order within a wave
        uint4 valq = *(const uint4*)&lds[wslot][m * 16 + half * 8];
        int row = (otile - 16) * 16 + m;      // value row in [0,512)
        *(uint4*)&hv[((size_t)(b * KO_ + row)) * N_ + ntile * 16 + half * 8] = valq;
    }
}

// ---------- 4) flash attention: one wave per (b, 16-query tile, 128-value segment) ----------
__global__ void k_attn(const _Float16* __restrict__ fT, const _Float16* __restrict__ gT,
                       const _Float16* __restrict__ hv, const float* __restrict__ gamma,
                       const float* __restrict__ mask, float* __restrict__ out) {
    int lane  = threadIdx.x & 31;
    int wave  = (blockIdx.x * blockDim.x + threadIdx.x) >> 5;
    int kseg  = wave & 3;
    int mtile = (wave >> 2) & 63;
    int b     = wave >> 8;
    int m = lane & 15, half = lane >> 4;
    int mcol = mtile * 16 + m;

    const uint4* F4 = (const uint4*)fT;    // row stride 16 uint4
    const uint4* G4 = (const uint4*)gT;    // row stride 16 uint4
    const uint4* H4 = (const uint4*)hv;    // row stride 2048 B = 128 uint4

    // Query operand (B matrix of scores GEMM), K=128 as 4 chunks of 32
    size_t qbase = ((size_t)(b * N_ + mcol)) * 16 + half;
    ABu qb[4];
#pragma unroll
    for (int chunk = 0; chunk < 4; ++chunk) {
        qb[chunk].q[0] = G4[qbase + chunk * 4];
        qb[chunk].q[1] = G4[qbase + chunk * 4 + 2];
    }

    Cu acc[8];
#pragma unroll
    for (int kt = 0; kt < 8; ++kt)
#pragma unroll
        for (int r = 0; r < 8; ++r) acc[kt].f[r] = 0.f;

    float mrun = -1e30f, lrun = 0.f;
    float gam  = gamma[0];

    for (int it = 0; it < 32; ++it) {   // 32 keys / iteration
        int n0 = it * 32;
        // prefetch next iteration's key rows (global_prefetch_b8)
        if (it < 31) {
            __builtin_prefetch((const void*)(F4 + ((size_t)(b * N_ + n0 + 32 + m)) * 16), 0, 3);
            __builtin_prefetch((const void*)(F4 + ((size_t)(b * N_ + n0 + 48 + m)) * 16), 0, 3);
        }
        Cu s[2];
#pragma unroll
        for (int t = 0; t < 2; ++t) {
#pragma unroll
            for (int r = 0; r < 8; ++r) s[t].f[r] = 0.f;
            size_t kbaseidx = ((size_t)(b * N_ + n0 + t * 16 + m)) * 16 + half;
#pragma unroll
            for (int chunk = 0; chunk < 4; ++chunk) {
                ABu a;
                a.q[0] = F4[kbaseidx + chunk * 4];
                a.q[1] = F4[kbaseidx + chunk * 4 + 2];
                s[t].v = __builtin_amdgcn_wmma_f32_16x16x32_f16(false, a.v, false, qb[chunk].v,
                                                                (short)0, s[t].v, false, false);
            }
        }
        // Online softmax over n (VGPR index + lane-half axis): per-lane reduce + one shfl_xor(16)
        float tmax = s[0].f[0];
#pragma unroll
        for (int t = 0; t < 2; ++t)
#pragma unroll
            for (int r = 0; r < 8; ++r) tmax = fmaxf(tmax, s[t].f[r]);
        tmax = fmaxf(tmax, __shfl_xor(tmax, 16));
        float mnew  = fmaxf(mrun, tmax);
        float scale = __expf(mrun - mnew);
        float rsum  = 0.f;
        ABu pb;      // score D-tile -> PV B-operand is a pure per-lane repack
#pragma unroll
        for (int t = 0; t < 2; ++t)
#pragma unroll
            for (int r = 0; r < 8; ++r) {
                float p = __expf(s[t].f[r] - mnew);
                rsum += p;
                pb.h[t * 8 + r] = (_Float16)p;
            }
        rsum += __shfl_xor(rsum, 16);
        lrun  = lrun * scale + rsum;
        mrun  = mnew;
#pragma unroll
        for (int kt = 0; kt < 8; ++kt)
#pragma unroll
            for (int r = 0; r < 8; ++r) acc[kt].f[r] *= scale;

        // PV: out[k, m] += hv[k, n0..n0+31] * P
#pragma unroll
        for (int kt = 0; kt < 8; ++kt) {
            ABu a;
            size_t hbase = ((size_t)(b * KO_ + kseg * 128 + kt * 16 + m)) * 128 + it * 4 + half;
            a.q[0] = H4[hbase];
            a.q[1] = H4[hbase + 2];
            acc[kt].v = __builtin_amdgcn_wmma_f32_16x16x32_f16(false, a.v, false, pb.v,
                                                               (short)0, acc[kt].v, false, false);
        }
    }

    float inv = gam / lrun;
#pragma unroll
    for (int kt = 0; kt < 8; ++kt)
#pragma unroll
        for (int r = 0; r < 8; ++r) {
            int k_abs  = kseg * 128 + kt * 16 + r + half * 8;
            int choice = k_abs >> 7;
            int oc     = k_abs & 127;
            size_t oi = (((size_t)(b * 128 + oc)) * 1024 + mcol) * 4 + choice;
            size_t mi = (((size_t)oc) * 1024 + mcol) * 4 + choice;
            out[oi] = acc[kt].f[r] * inv + mask[mi];
        }
}

extern "C" void kernel_launch(void* const* d_in, const int* in_sizes, int n_in,
                              void* d_out, int out_size, void* d_ws, size_t ws_size,
                              hipStream_t stream) {
    (void)in_sizes; (void)n_in; (void)out_size; (void)ws_size;
    const float* x     = (const float*)d_in[0];
    const float* Wq    = (const float*)d_in[1];
    const float* Wk    = (const float*)d_in[2];
    const float* Wv    = (const float*)d_in[3];
    const float* gamma = (const float*)d_in[4];
    const float* mask  = (const float*)d_in[5];
    float* out = (float*)d_out;

    char* ws = (char*)d_ws;
    _Float16* xT  = (_Float16*)(ws);                          // 4 MiB  [B,N,C]
    _Float16* W16 = (_Float16*)(ws + (4u << 20));             // 192 KiB [768,128]
    _Float16* fT  = (_Float16*)(ws + (5u << 20));             // 4 MiB  [B,N,128]
    _Float16* gT  = (_Float16*)(ws + (9u << 20));             // 4 MiB  [B,N,128]
    _Float16* hv  = (_Float16*)(ws + (13u << 20));            // 16 MiB [B,512,N]

    // transpose-convert: B * (C/32) * (N/32) = 2048 blocks of 32x8
    k_convert_x<<<2048, dim3(32, 8), 0, stream>>>(x, xT);
    k_convert_w<<<(PR_ * C_) / 256, 256, 0, stream>>>(Wq, Wk, Wv, W16);
    // tiles: B * (768/16) * (1024/16) = 49152 waves, 8 waves/block
    k_proj<<<49152 / 8, 256, 0, stream>>>(W16, xT, fT, gT, hv);
    // waves: B * 64 mtiles * 4 ksegs = 4096, 8 waves/block
    k_attn<<<4096 / 8, 256, 0, stream>>>(fT, gT, hv, gamma, mask, out);
}
